// Attention_50861002719734
// MI455X (gfx1250) — compile-verified, matrix operand order checked by reading, not run on hardware
//
#include <hip/hip_runtime.h>
#include <math.h>

// ---------------------------------------------------------------------------
// Causal attention (flash-attention style) for MI455X / gfx1250, wave32 WMMA.
// B=16, L=2048, D=1024, fp32 in/out; f16 WMMA with fp32 accumulation.
// QT=32 query rows per block; V staged transposed in LDS so both GEMMs use
// 16B ds_load_b128 operands and fully coalesced global traffic.
// ---------------------------------------------------------------------------

typedef __attribute__((ext_vector_type(16))) _Float16 v16h;
typedef __attribute__((ext_vector_type(8)))  _Float16 h8;
typedef __attribute__((ext_vector_type(4)))  _Float16 h4;
typedef __attribute__((ext_vector_type(8)))  float    v8f;
typedef __attribute__((ext_vector_type(4)))  float    f32x4;

#define BATCH   16
#define SEQ     2048
#define DMODEL  1024
#define QT      32          // query rows per block (2 M-tiles of 16)
#define KVT     128         // kv rows per block-iteration
#define VROWS   64          // V staging slab (2 slabs per KV block)
#define VTS     72          // padded k-stride of Vt (halves): 16B-aligned reads,
                            // conflict-free 8B stores (36-dword bank stride)
#define THREADS 256         // 8 waves of 32
#define NEG_BIG (-3.0e38f)
// softmax uses exp((s-m)/32) = exp2((s-m) * 1/(32*ln2))
#define KSCALE  0.0450842306f

__global__ __launch_bounds__(THREADS)
void fa_causal_wmma(const float* __restrict__ Q,
                    const float* __restrict__ K,
                    const float* __restrict__ V,
                    float* __restrict__ O)
{
    // ---- static LDS (~234 KB of the 320 KB CDNA5 WGP budget)
    __shared__ _Float16 Qs[QT * DMODEL];     //  64 KB, Q tile f16
    __shared__ _Float16 Vt[DMODEL * VTS];    // 144 KB, V slab transposed [d][k]
    __shared__ float    Ss[QT * KVT];        //  16 KB, raw scores
    __shared__ _Float16 Ps[QT * KVT];        //   8 KB, probabilities f16
    __shared__ float    Pm[QT * 8];          //   1 KB, partial row maxes
    __shared__ float    Sm[QT * 8];          //   1 KB, partial row sums
    __shared__ float    Ms[QT];              // running max
    __shared__ float    Ls[QT];              // running sum
    __shared__ float    Cs[QT];              // per-round rescale factor

    const int qb   = blockIdx.x;          // query tile index (0..63)
    const int b    = blockIdx.y;          // batch
    const int tid  = threadIdx.x;
    const int wave = tid >> 5;
    const int lane = tid & 31;
    const int m    = lane & 15;           // row (A) / col (B,C) within 16
    const int koff = (lane >> 4) * 8;     // k-offset from lane half (ISA layout)

    const float* Qb = Q + ((size_t)b * SEQ + (size_t)qb * QT) * DMODEL;
    const float* Kb = K + (size_t)b * SEQ * DMODEL;
    const float* Vb = V + (size_t)b * SEQ * DMODEL;

    // ---- stage Q tile to LDS as f16 (coalesced float4 loads, 8B LDS stores)
    for (int i = tid * 4; i < QT * DMODEL; i += THREADS * 4) {
        f32x4 q4 = *(const f32x4*)(Qb + i);
        h4 t;
        t[0] = (_Float16)q4.x; t[1] = (_Float16)q4.y;
        t[2] = (_Float16)q4.z; t[3] = (_Float16)q4.w;
        *(h4*)(Qs + i) = t;
    }
    if (tid < QT) { Ms[tid] = NEG_BIG; Ls[tid] = 0.0f; }
    __syncthreads();

    // ---- per-wave accumulators: 2 M-tiles x 8 N-tiles (D cols wave*128..+128)
    v8f acc[2][8];
    v8f zero = {};
#pragma unroll
    for (int mt = 0; mt < 2; ++mt)
#pragma unroll
        for (int t = 0; t < 8; ++t) acc[mt][t] = zero;

    const int qlast = qb * QT + (QT - 1);
    const int nkv   = qlast / KVT + 1;    // uniform per block (causal bound)

    for (int j = 0; j < nkv; ++j) {
        const int kv0 = j * KVT;

        // prefetch next KV block into GL2 (global_prefetch_b8)
        if (j + 1 < nkv) {
            __builtin_prefetch(Kb + (size_t)(kv0 + KVT) * DMODEL + tid * 16, 0, 1);
            __builtin_prefetch(Vb + (size_t)(kv0 + KVT) * DMODEL + tid * 16, 0, 1);
        }

        // ============ Phase 1: S = Q (32xD) * K^T for this wave's 16 kv cols.
        // One K chunk (B operand) feeds both M-tiles -> 2 WMMAs per 8 cvt_pk.
        {
            v8f s0 = zero, s1 = zero;
            const float* Krow = Kb + (size_t)(kv0 + wave * 16 + m) * DMODEL;
            for (int dc = 0; dc < DMODEL; dc += 32) {
                // B = K^T: lane = kv column, halves = d (contiguous f32 loads)
                f32x4 x0 = *(const f32x4*)(Krow + dc + koff);
                f32x4 x1 = *(const f32x4*)(Krow + dc + koff + 4);
                f32x4 y0 = *(const f32x4*)(Krow + dc + koff + 16);
                f32x4 y1 = *(const f32x4*)(Krow + dc + koff + 20);
                v16h bk;
                bk[0]=(_Float16)x0.x; bk[1]=(_Float16)x0.y; bk[2]=(_Float16)x0.z; bk[3]=(_Float16)x0.w;
                bk[4]=(_Float16)x1.x; bk[5]=(_Float16)x1.y; bk[6]=(_Float16)x1.z; bk[7]=(_Float16)x1.w;
                bk[8]=(_Float16)y0.x; bk[9]=(_Float16)y0.y; bk[10]=(_Float16)y0.z; bk[11]=(_Float16)y0.w;
                bk[12]=(_Float16)y1.x; bk[13]=(_Float16)y1.y; bk[14]=(_Float16)y1.z; bk[15]=(_Float16)y1.w;
                // A = Q[mt*16+m, dc..dc+32) from LDS (two 16B ds reads each)
                const _Float16* ap0 = Qs + m * DMODEL + dc + koff;
                const _Float16* ap1 = Qs + (16 + m) * DMODEL + dc + koff;
                h8 lo0 = *(const h8*)ap0, hi0 = *(const h8*)(ap0 + 16);
                h8 lo1 = *(const h8*)ap1, hi1 = *(const h8*)(ap1 + 16);
                v16h a0 = __builtin_shufflevector(lo0, hi0,
                        0,1,2,3,4,5,6,7,8,9,10,11,12,13,14,15);
                v16h a1 = __builtin_shufflevector(lo1, hi1,
                        0,1,2,3,4,5,6,7,8,9,10,11,12,13,14,15);
                s0 = __builtin_amdgcn_wmma_f32_16x16x32_f16(
                        false, a0, false, bk, (short)0, s0, false, false);
                s1 = __builtin_amdgcn_wmma_f32_16x16x32_f16(
                        false, a1, false, bk, (short)0, s1, false, false);
            }
            // scatter fp32 scores (C layout: vgpr r -> row r + 8*(lane/16))
#pragma unroll
            for (int r = 0; r < 8; ++r) {
                const int row = r + 8 * (lane >> 4);
                Ss[row * KVT + wave * 16 + m]        = s0[r];
                Ss[(16 + row) * KVT + wave * 16 + m] = s1[r];
            }
        }
        __syncthreads();

        // ============ Phase 2: online softmax, parallel over all 256 threads
        {
            const int row = tid >> 3;         // 0..31
            const int seg = tid & 7;          // 0..7, 16 cols each
            const int qg  = qb * QT + row;
            const int c0  = seg * 16;
            float pm = NEG_BIG;
            for (int c = c0; c < c0 + 16; ++c) {
                const bool ok = (kv0 + c) <= qg;          // causal mask
                const float x = ok ? Ss[row * KVT + c] : NEG_BIG;
                pm = fmaxf(pm, x);
            }
            Pm[row * 8 + seg] = pm;
        }
        __syncthreads();
        if (tid < QT) {
            const float m_old = Ms[tid];
            float bm = NEG_BIG;
#pragma unroll
            for (int s = 0; s < 8; ++s) bm = fmaxf(bm, Pm[tid * 8 + s]);
            const float m_new = fmaxf(m_old, bm);
            Cs[tid] = exp2f((m_old - m_new) * KSCALE);
            Ms[tid] = m_new;
        }
        __syncthreads();
        {
            const int row = tid >> 3;
            const int seg = tid & 7;
            const int qg  = qb * QT + row;
            const int c0  = seg * 16;
            const float mn = Ms[row];
            float ls = 0.0f;
            for (int c = c0; c < c0 + 16; ++c) {
                const bool ok = (kv0 + c) <= qg;
                float p = 0.0f;
                if (ok) p = exp2f((Ss[row * KVT + c] - mn) * KSCALE);
                Ps[row * KVT + c] = (_Float16)p;
                ls += p;
            }
            Sm[row * 8 + seg] = ls;
        }
        __syncthreads();
        if (tid < QT) {
            float ls = 0.0f;
#pragma unroll
            for (int s = 0; s < 8; ++s) ls += Sm[tid * 8 + s];
            Ls[tid] = Ls[tid] * Cs[tid] + ls;
        }
        __syncthreads();

        // ============ Phase 3: rescale accumulators, then O += P * V
#pragma unroll
        for (int mt = 0; mt < 2; ++mt)
#pragma unroll
            for (int r = 0; r < 8; ++r) {
                const float cr = Cs[mt * 16 + r + 8 * (lane >> 4)];
#pragma unroll
                for (int t = 0; t < 8; ++t) acc[mt][t][r] *= cr;
            }

        for (int half = 0; half < 2; ++half) {     // two 64-row V slabs
            __syncthreads();                       // Vt WAR protection
            // ---- stage V slab transposed: Vt[d*VTS + k] = V[kv0+half*64+k][d]
            // per k-row, the 256 threads' float4 loads cover exactly the row.
            const float* Vc = Vb + (size_t)(kv0 + half * VROWS) * DMODEL;
            if (half == 0)   // prefetch second slab while staging first
                __builtin_prefetch(Vc + (size_t)VROWS * DMODEL + tid * 16, 0, 1);
            for (int kq = 0; kq < VROWS; kq += 4) {
                const float* p = Vc + (size_t)kq * DMODEL + tid * 4;
                f32x4 r0 = *(const f32x4*)(p);
                f32x4 r1 = *(const f32x4*)(p + DMODEL);
                f32x4 r2 = *(const f32x4*)(p + 2 * DMODEL);
                f32x4 r3 = *(const f32x4*)(p + 3 * DMODEL);
                h4 c0, c1, c2, c3;   // 4x4 register transpose + f16 convert
                c0[0]=(_Float16)r0.x; c0[1]=(_Float16)r1.x; c0[2]=(_Float16)r2.x; c0[3]=(_Float16)r3.x;
                c1[0]=(_Float16)r0.y; c1[1]=(_Float16)r1.y; c1[2]=(_Float16)r2.y; c1[3]=(_Float16)r3.y;
                c2[0]=(_Float16)r0.z; c2[1]=(_Float16)r1.z; c2[2]=(_Float16)r2.z; c2[3]=(_Float16)r3.z;
                c3[0]=(_Float16)r0.w; c3[1]=(_Float16)r1.w; c3[2]=(_Float16)r2.w; c3[3]=(_Float16)r3.w;
                *(h4*)(Vt + (size_t)(tid * 4 + 0) * VTS + kq) = c0;
                *(h4*)(Vt + (size_t)(tid * 4 + 1) * VTS + kq) = c1;
                *(h4*)(Vt + (size_t)(tid * 4 + 2) * VTS + kq) = c2;
                *(h4*)(Vt + (size_t)(tid * 4 + 3) * VTS + kq) = c3;
            }
            __syncthreads();

            // ---- consume slab: 2 k-slabs of 32, B operands now 2x ds_load_b128
            for (int kc = 0; kc < 2; ++kc) {
                const int kcol = half * VROWS + kc * 32;   // column base in Ps
                const _Float16* pp0 = Ps + m * KVT + kcol + koff;
                const _Float16* pp1 = Ps + (16 + m) * KVT + kcol + koff;
                h8 pl0 = *(const h8*)pp0, ph0 = *(const h8*)(pp0 + 16);
                h8 pl1 = *(const h8*)pp1, ph1 = *(const h8*)(pp1 + 16);
                v16h pa0 = __builtin_shufflevector(pl0, ph0,
                        0,1,2,3,4,5,6,7,8,9,10,11,12,13,14,15);
                v16h pa1 = __builtin_shufflevector(pl1, ph1,
                        0,1,2,3,4,5,6,7,8,9,10,11,12,13,14,15);
#pragma unroll
                for (int t = 0; t < 8; ++t) {
                    // B = V^T slab: lane = D column, halves = kv row (contiguous)
                    const int dcol = wave * 128 + t * 16 + m;
                    const _Float16* vp = Vt + (size_t)dcol * VTS + kc * 32 + koff;
                    h8 vlo = *(const h8*)vp;
                    h8 vhi = *(const h8*)(vp + 16);
                    v16h vbv = __builtin_shufflevector(vlo, vhi,
                            0,1,2,3,4,5,6,7,8,9,10,11,12,13,14,15);
                    acc[0][t] = __builtin_amdgcn_wmma_f32_16x16x32_f16(
                            false, pa0, false, vbv, (short)0, acc[0][t], false, false);
                    acc[1][t] = __builtin_amdgcn_wmma_f32_16x16x32_f16(
                            false, pa1, false, vbv, (short)0, acc[1][t], false, false);
                }
            }
        }
    }

    // ============ Epilogue: O = acc / l_row
#pragma unroll
    for (int mt = 0; mt < 2; ++mt)
#pragma unroll
        for (int r = 0; r < 8; ++r) {
            const int   row = mt * 16 + r + 8 * (lane >> 4);
            const float inv = 1.0f / Ls[row];
#pragma unroll
            for (int t = 0; t < 8; ++t) {
                const int dcol = wave * 128 + t * 16 + m;
                O[((size_t)b * SEQ + (size_t)qb * QT + row) * DMODEL + dcol]
                    = acc[mt][t][r] * inv;
            }
        }
}

extern "C" void kernel_launch(void* const* d_in, const int* in_sizes, int n_in,
                              void* d_out, int out_size, void* d_ws, size_t ws_size,
                              hipStream_t stream) {
    (void)in_sizes; (void)n_in; (void)out_size; (void)d_ws; (void)ws_size;
    const float* Q = (const float*)d_in[0];
    const float* K = (const float*)d_in[1];
    const float* V = (const float*)d_in[2];
    float*       O = (float*)d_out;
    dim3 grid(SEQ / QT, BATCH);
    fa_causal_wmma<<<grid, THREADS, 0, stream>>>(Q, K, V, O);
}
